// SparseSelfTransformerBlock_69346541961599
// MI455X (gfx1250) — matrix operand order repaired and checked by reading, not compiled, not run on hardware
//
#include <hip/hip_runtime.h>
#include <math.h>

// ---------------------------------------------------------------------------
// Types for CDNA5 WMMA (wave32): bf16 16x16x32, f32 accumulate
// ---------------------------------------------------------------------------
typedef __attribute__((ext_vector_type(16))) __bf16 v16bf;
typedef __attribute__((ext_vector_type(2)))  __bf16 v2bf;
typedef __attribute__((ext_vector_type(8)))  float  v8f;

// Fixed problem dims from the reference
#define BATCH 2
#define SEQ   2048
#define DMODEL 512
#define HEADS 8
#define HDIM  64
#define MROWS (BATCH * SEQ)   // 4096
#define SPAN  64

// ---------------------------------------------------------------------------
// Fragment loaders (CDNA5 ISA 7.12.2 layouts). The compiler coalesces these
// into 2x ds_load_b128 per fragment (verified in round-1 disasm).
// A: 16x32 bf16, source LDS row-major [m][k], leading dim `ld` (halves).
// B: 32x16 bf16 (K x N) where B[k][n] = W[n][k]; source holds W row-major [n][k].
// ---------------------------------------------------------------------------
__device__ __forceinline__ v16bf load_frag_a(const __bf16* base, int ld) {
  const int lane = threadIdx.x & 31;
  const int m  = lane & 15;
  const int kh = (lane >> 4) & 1;
  v16bf a;
#pragma unroll
  for (int v = 0; v < 8; ++v) {
    const int kb = (v < 4 ? 2 * v : 16 + 2 * (v - 4)) + 8 * kh;
    v2bf p = *(const v2bf*)(base + m * ld + kb);
    a[2 * v]     = p[0];
    a[2 * v + 1] = p[1];
  }
  return a;
}

__device__ __forceinline__ v16bf load_frag_b(const __bf16* base, int ld) {
  const int lane = threadIdx.x & 31;
  const int n  = lane & 15;
  const int kh = (lane >> 4) & 1;
  v16bf b;
#pragma unroll
  for (int v = 0; v < 8; ++v) {
    const int kb = (v < 4 ? 2 * v : 16 + 2 * (v - 4)) + 8 * kh;
    v2bf p = *(const v2bf*)(base + n * ld + kb);
    b[2 * v]     = p[0];
    b[2 * v + 1] = p[1];
  }
  return b;
}

// ---------------------------------------------------------------------------
// f32 -> bf16 conversion (weights)
// ---------------------------------------------------------------------------
__global__ __launch_bounds__(256)
void cvt_f32_bf16(const float* __restrict__ src, __bf16* __restrict__ dst, int n) {
  int i = blockIdx.x * 256 + threadIdx.x;
  if (i < n) dst[i] = (__bf16)src[i];
}

// ---------------------------------------------------------------------------
// LayerNorm: one block (128 thr, 4 waves) per row of 512; outputs bf16
// ---------------------------------------------------------------------------
__global__ __launch_bounds__(128)
void layernorm_bf16(const float* __restrict__ x, const float* __restrict__ g,
                    const float* __restrict__ bta, __bf16* __restrict__ out) {
  const int row  = blockIdx.x;
  const int t    = threadIdx.x;
  const int lane = t & 31, wave = t >> 5;
  const float* xr = x + (size_t)row * DMODEL;

  float4 f = ((const float4*)xr)[t];           // 128 thr * 4 = 512
  float s  = f.x + f.y + f.z + f.w;
  float s2 = f.x * f.x + f.y * f.y + f.z * f.z + f.w * f.w;
#pragma unroll
  for (int o = 16; o >= 1; o >>= 1) {
    s  += __shfl_xor(s,  o, 32);
    s2 += __shfl_xor(s2, o, 32);
  }
  __shared__ float red[10];
  if (lane == 0) { red[wave] = s; red[4 + wave] = s2; }
  __syncthreads();
  if (t == 0) {
    float ts  = red[0] + red[1] + red[2] + red[3];
    float ts2 = red[4] + red[5] + red[6] + red[7];
    float mu  = ts * (1.0f / DMODEL);
    float var = ts2 * (1.0f / DMODEL) - mu * mu;
    red[8] = mu;
    red[9] = rsqrtf(var + 1e-5f);
  }
  __syncthreads();
  const float mu = red[8], inv = red[9];
  const float vv[4] = {f.x, f.y, f.z, f.w};
#pragma unroll
  for (int i = 0; i < 4; ++i) {
    int c = t * 4 + i;
    out[(size_t)row * DMODEL + c] = (__bf16)((vv[i] - mu) * inv * g[c] + bta[c]);
  }
}

// ---------------------------------------------------------------------------
// WMMA GEMM: C[M x N] (f32 acc) = A[M x K](bf16,row-major) * W[N x K]^T
// 128x64 tile, 256 threads (8 waves, one 16-row strip each), K-step 32.
// Software-pipelined: global loads for step k+1 issue into registers while
// WMMAs consume LDS buffer of step k; double-buffered LDS, one barrier/step.
// Epilogue is a compile-time template (no runtime branches in store loop).
// ---------------------------------------------------------------------------
enum { EPI_BF16 = 0, EPI_RES_F32 = 1, EPI_BIAS_GELU_BF16 = 2, EPI_BIAS_RES_F32 = 3 };

template <int MODE>
__global__ __launch_bounds__(256)
void gemm_bf16_wmma(const __bf16* __restrict__ A, const __bf16* __restrict__ W,
                    int K, int N,
                    const float* __restrict__ bias,
                    const float* __restrict__ resid,
                    float* __restrict__ outf, __bf16* __restrict__ outb) {
  __shared__ __bf16 As[2][128][40];   // 80B rows: 16B-aligned, conflict-friendly
  __shared__ __bf16 Ws[2][64][40];
  const int t = threadIdx.x, lane = t & 31, wave = t >> 5;
  const int m0 = blockIdx.y * 128, n0 = blockIdx.x * 64;

  // Per-thread staging map: 16B chunks. A: 128 rows x 4 chunks = 512 (2/thread),
  // W: 64 rows x 4 chunks = 256 (1/thread).
  const int ar0 = t >> 2,           ac0 = (t & 3) * 8;
  const int ar1 = (t + 256) >> 2,   ac1 = ac0;
  const int wr  = t >> 2,           wc  = (t & 3) * 8;

  uint4 ra0 = *(const uint4*)(A + (size_t)(m0 + ar0) * K + ac0);
  uint4 ra1 = *(const uint4*)(A + (size_t)(m0 + ar1) * K + ac1);
  uint4 rw  = *(const uint4*)(W + (size_t)(n0 + wr)  * K + wc);

  v8f acc[4] = {};
  int buf = 0;

  for (int kk = 0; kk < K; kk += 32) {
    // commit staged registers into current LDS buffer
    *(uint4*)&As[buf][ar0][ac0] = ra0;
    *(uint4*)&As[buf][ar1][ac1] = ra1;
    *(uint4*)&Ws[buf][wr][wc]   = rw;
    __syncthreads();

    // issue next slab's global loads (overlap with WMMA below)
    if (kk + 32 < K) {
      ra0 = *(const uint4*)(A + (size_t)(m0 + ar0) * K + kk + 32 + ac0);
      ra1 = *(const uint4*)(A + (size_t)(m0 + ar1) * K + kk + 32 + ac1);
      rw  = *(const uint4*)(W + (size_t)(n0 + wr)  * K + kk + 32 + wc);
    }

    v16bf af = load_frag_a(&As[buf][wave * 16][0], 40);
#pragma unroll
    for (int nt = 0; nt < 4; ++nt) {
      v16bf wf = load_frag_b(&Ws[buf][nt * 16][0], 40);
      acc[nt] = __builtin_amdgcn_wmma_f32_16x16x32_bf16(
          false, af, false, wf, (short)0, acc[nt], false, false);
    }
    buf ^= 1;
  }

#pragma unroll
  for (int nt = 0; nt < 4; ++nt) {
#pragma unroll
    for (int r = 0; r < 8; ++r) {
      const int row = m0 + wave * 16 + r + 8 * (lane >> 4);
      const int col = n0 + nt * 16 + (lane & 15);
      const size_t idx = (size_t)row * N + col;
      const float v = acc[nt][r];
      if (MODE == EPI_BF16) {
        outb[idx] = (__bf16)v;
      } else if (MODE == EPI_RES_F32) {
        outf[idx] = v + resid[idx];
      } else if (MODE == EPI_BIAS_GELU_BF16) {
        float u = v + bias[col];
        outb[idx] = (__bf16)(0.5f * u * (1.0f + erff(u * 0.70710678118f)));
      } else { // EPI_BIAS_RES_F32
        outf[idx] = v + bias[col] + resid[idx];
      }
    }
  }
}

// ---------------------------------------------------------------------------
// Windowed causal attention (span=64, stride=1).
// One block per (b, h, 64-query tile); 128 threads = 4 waves, wave w owns
// query rows [w*16, w*16+16). Keys for the tile live in [qbase-64, qbase+64).
// qkv layout: [B*L][1536] bf16, cols 0..511=Q, 512..1023=K, 1024..1535=V.
// Softmax in registers over the WMMA C-fragment layout, wave32 shfl_xor.
// ---------------------------------------------------------------------------
__global__ __launch_bounds__(128)
void window_attn_wmma(const __bf16* __restrict__ qkv, __bf16* __restrict__ attn) {
  const int qt = blockIdx.x, h = blockIdx.y, b = blockIdx.z;
  const int qbase = qt * 64, kbase = qbase - 64;
  const int t = threadIdx.x, lane = t & 31, wave = t >> 5;
  const int nl = lane & 15, hh = lane >> 4;

  __shared__ __bf16 Qs[64][72];     // [q][d]
  __shared__ __bf16 Ks[128][72];    // [k][d]
  __shared__ __bf16 Vt[64][136];    // [d][k]  (V transposed -> B-operand layout)
  __shared__ __bf16 Ps[64][136];    // [q][k]  (probabilities, wave-private rows)

  // ---- stage Q (64x64), K (128x64), V^T (64x128) ----
#pragma unroll
  for (int i = 0; i < 4; ++i) {
    int c = t + i * 128, row = c >> 3, col = (c & 7) * 8;
    *(uint4*)&Qs[row][col] =
        *(const uint4*)(qkv + (size_t)(b * SEQ + qbase + row) * 1536 + h * HDIM + col);
  }
#pragma unroll
  for (int i = 0; i < 8; ++i) {
    int c = t + i * 128, row = c >> 3, col = (c & 7) * 8;
    int gk = kbase + row; if (gk < 0) gk = 0;   // clamped; masked in scores
    *(uint4*)&Ks[row][col] =
        *(const uint4*)(qkv + (size_t)(b * SEQ + gk) * 1536 + 512 + h * HDIM + col);
  }
#pragma unroll
  for (int i = 0; i < 8; ++i) {
    int c = t + i * 128, j = c >> 3, col = (c & 7) * 8;
    int gk = kbase + j; if (gk < 0) gk = 0;
    const __bf16* src = qkv + (size_t)(b * SEQ + gk) * 1536 + 1024 + h * HDIM + col;
#pragma unroll
    for (int e = 0; e < 8; ++e) Vt[col + e][j] = src[e];
  }
  __syncthreads();

  // ---- S = Q * K^T : 8 key tiles, K-dim 64 (2 steps) ----
  v8f S[8] = {};
#pragma unroll
  for (int ks = 0; ks < 2; ++ks) {
    v16bf af = load_frag_a(&Qs[wave * 16][ks * 32], 72);
#pragma unroll
    for (int kt = 0; kt < 8; ++kt) {
      v16bf bf = load_frag_b(&Ks[kt * 16][ks * 32], 72);
      S[kt] = __builtin_amdgcn_wmma_f32_16x16x32_bf16(
          false, af, false, bf, (short)0, S[kt], false, false);
    }
  }

  // ---- mask + scale + softmax in registers ----
  const float scale = 0.125f; // 1/sqrt(64)
#pragma unroll
  for (int r = 0; r < 8; ++r) {
    const int qpos = qbase + wave * 16 + r + 8 * hh;
    float mx = -1e30f;
#pragma unroll
    for (int kt = 0; kt < 8; ++kt) {
      const int kpos = kbase + kt * 16 + nl;
      const bool valid = (kpos >= 0) && (kpos <= qpos) && (kpos >= qpos - (SPAN - 1));
      float v = valid ? S[kt][r] * scale : -1e30f;
      S[kt][r] = v;
      mx = fmaxf(mx, v);
    }
#pragma unroll
    for (int o = 8; o >= 1; o >>= 1) mx = fmaxf(mx, __shfl_xor(mx, o, 32));
    float sum = 0.f;
#pragma unroll
    for (int kt = 0; kt < 8; ++kt) {
      float e = __expf(S[kt][r] - mx);
      S[kt][r] = e;
      sum += e;
    }
#pragma unroll
    for (int o = 8; o >= 1; o >>= 1) sum += __shfl_xor(sum, o, 32);
    const float inv = 1.0f / sum;
    const int qrl = wave * 16 + r + 8 * hh;
#pragma unroll
    for (int kt = 0; kt < 8; ++kt)
      Ps[qrl][kt * 16 + nl] = (__bf16)(S[kt][r] * inv);
  }
  __syncthreads();   // rows are wave-private; barrier for clean LDS ordering

  // ---- O = P * V : K-dim 128 (4 steps), 4 d-tiles of 16 ----
  v8f O[4] = {};
#pragma unroll
  for (int ks = 0; ks < 4; ++ks) {
    v16bf af = load_frag_a(&Ps[wave * 16][ks * 32], 136);
#pragma unroll
    for (int nt = 0; nt < 4; ++nt) {
      v16bf bf = load_frag_b(&Vt[nt * 16][ks * 32], 136);
      O[nt] = __builtin_amdgcn_wmma_f32_16x16x32_bf16(
          false, af, false, bf, (short)0, O[nt], false, false);
    }
  }

  // ---- store: attn[b][q][h*64+dd] bf16 ----
#pragma unroll
  for (int nt = 0; nt < 4; ++nt) {
#pragma unroll
    for (int r = 0; r < 8; ++r) {
      const int q  = qbase + wave * 16 + r + 8 * hh;
      const int dd = nt * 16 + nl;
      attn[(size_t)(b * SEQ + q) * DMODEL + h * HDIM + dd] = (__bf16)O[nt][r];
    }
  }
}

// ---------------------------------------------------------------------------
// Host side
// ---------------------------------------------------------------------------
extern "C" void kernel_launch(void* const* d_in, const int* in_sizes, int n_in,
                              void* d_out, int out_size, void* d_ws, size_t ws_size,
                              hipStream_t stream) {
  (void)in_sizes; (void)n_in; (void)out_size; (void)ws_size;
  const float* query = (const float*)d_in[0];
  const float* ln_g  = (const float*)d_in[1];
  const float* ln_b  = (const float*)d_in[2];
  const float* wq    = (const float*)d_in[3];
  const float* wk    = (const float*)d_in[4];
  const float* wv    = (const float*)d_in[5];
  const float* wo    = (const float*)d_in[6];
  const float* w1    = (const float*)d_in[7];
  const float* b1    = (const float*)d_in[8];
  const float* w2    = (const float*)d_in[9];
  const float* b2    = (const float*)d_in[10];
  float* out = (float*)d_out;

  char* ws = (char*)d_ws;
  size_t off = 0;
  auto alloc = [&](size_t bytes) -> void* {
    void* p = ws + off;
    off += (bytes + 255) & ~(size_t)255;
    return p;
  };
  __bf16* wqkv_b = (__bf16*)alloc((size_t)1536 * 512 * 2);
  __bf16* wo_b   = (__bf16*)alloc((size_t)512 * 512 * 2);
  __bf16* w1_b   = (__bf16*)alloc((size_t)1024 * 512 * 2);
  __bf16* w2_b   = (__bf16*)alloc((size_t)512 * 1024 * 2);
  __bf16* xln    = (__bf16*)alloc((size_t)MROWS * 512 * 2);
  __bf16* qkv    = (__bf16*)alloc((size_t)MROWS * 1536 * 2);
  __bf16* attnb  = (__bf16*)alloc((size_t)MROWS * 512 * 2);
  float*  y1     = (float*)alloc((size_t)MROWS * 512 * 4);
  __bf16* xln2   = (__bf16*)alloc((size_t)MROWS * 512 * 2);
  __bf16* hmid   = (__bf16*)alloc((size_t)MROWS * 1024 * 2);

  const int nW = 512 * 512;     // 262144
  // pack wq|wk|wv -> wqkv (rows 0..511 | 512..1023 | 1024..1535)
  cvt_f32_bf16<<<(nW + 255) / 256, 256, 0, stream>>>(wq, wqkv_b,          nW);
  cvt_f32_bf16<<<(nW + 255) / 256, 256, 0, stream>>>(wk, wqkv_b + nW,     nW);
  cvt_f32_bf16<<<(nW + 255) / 256, 256, 0, stream>>>(wv, wqkv_b + 2 * nW, nW);
  cvt_f32_bf16<<<(nW + 255) / 256, 256, 0, stream>>>(wo, wo_b,            nW);
  cvt_f32_bf16<<<(2 * nW + 255) / 256, 256, 0, stream>>>(w1, w1_b, 2 * nW);
  cvt_f32_bf16<<<(2 * nW + 255) / 256, 256, 0, stream>>>(w2, w2_b, 2 * nW);

  // LN1
  layernorm_bf16<<<MROWS, 128, 0, stream>>>(query, ln_g, ln_b, xln);

  // QKV projection: [4096x512] * [1536x512]^T -> bf16 qkv
  gemm_bf16_wmma<EPI_BF16><<<dim3(1536 / 64, MROWS / 128), 256, 0, stream>>>(
      xln, wqkv_b, 512, 1536, nullptr, nullptr, nullptr, qkv);

  // windowed attention
  window_attn_wmma<<<dim3(SEQ / 64, HEADS, BATCH), 128, 0, stream>>>(qkv, attnb);

  // output projection + residual: y1 = attn @ wo^T + query
  gemm_bf16_wmma<EPI_RES_F32><<<dim3(512 / 64, MROWS / 128), 256, 0, stream>>>(
      attnb, wo_b, 512, 512, nullptr, query, y1, nullptr);

  // LN2
  layernorm_bf16<<<MROWS, 128, 0, stream>>>(y1, ln_g, ln_b, xln2);

  // FFN1: gelu(xln2 @ w1^T + b1) -> bf16
  gemm_bf16_wmma<EPI_BIAS_GELU_BF16><<<dim3(1024 / 64, MROWS / 128), 256, 0, stream>>>(
      xln2, w1_b, 512, 1024, b1, nullptr, nullptr, hmid);

  // FFN2: out = hmid @ w2^T + b2 + y1
  gemm_bf16_wmma<EPI_BIAS_RES_F32><<<dim3(512 / 64, MROWS / 128), 256, 0, stream>>>(
      hmid, w2_b, 1024, 512, b2, y1, out, nullptr);
}
